// CausalSelfAttention_19920058319239
// MI455X (gfx1250) — matrix operand order repaired and checked by reading, not compiled, not run on hardware
//
#include <hip/hip_runtime.h>
#include <hip/hip_bf16.h>

typedef __attribute__((ext_vector_type(16))) __bf16 v16bf;
typedef __attribute__((ext_vector_type(8)))  float  v8f;

#define BATCH 4
#define SEQ   2048
#define CH    1024
#define NHEAD 16
#define HDIM  64
#define C3    (3 * CH)
#define C2    (2 * CH)

union Frag16 { v16bf v; uint4 q[2]; };

// LDS flat address low 32 bits == workgroup-relative LDS byte offset (ISA 10.2)
__device__ inline unsigned lds_off(const void* p) {
    return (unsigned)(unsigned long long)p;
}

// CDNA5 async global->LDS 16B copy, tracked by ASYNCcnt (ISA 10 / 15.18.3 op 98)
__device__ inline void async_cp16(unsigned lds, const void* g) {
    asm volatile("global_load_async_to_lds_b128 %0, %1, off"
                 :: "v"(lds), "v"(g) : "memory");
}
__device__ inline void wait_async0() {
#if __has_builtin(__builtin_amdgcn_s_wait_asynccnt)
    __builtin_amdgcn_s_wait_asynccnt(0);
#else
    asm volatile("s_wait_asynccnt 0" ::: "memory");
#endif
}

// A-matrix 16x32 bf16 fragment (ISA 7.12.2): lane m = lane%16,
// elements 0..7 -> K = 8*hi.. ; elements 8..15 -> K = 16+8*hi..
__device__ inline v16bf load_afrag(const __bf16* row, int hi) {
    Frag16 f;
    f.q[0] = *(const uint4*)(row + 8 * hi);
    f.q[1] = *(const uint4*)(row + 16 + 8 * hi);
    return f.v;
}
// B-matrix 32x16 bf16 fragment: lane n = lane%16, elements 0..7 -> K = 8*hi.. ; elements 8..15 -> K = 16+8*hi..
__device__ inline v16bf load_bfrag(const __bf16* row, int hi) {
    Frag16 f;
    f.q[0] = *(const uint4*)(row + 8 * hi);
    f.q[1] = *(const uint4*)(row + 16 + 8 * hi);
    return f.v;
}
__device__ inline v8f wmma_bf16(v16bf a, v16bf b, v8f c) {
    return __builtin_amdgcn_wmma_f32_16x16x32_bf16(
        false, a, false, b, (short)0, c, false, false);
}

// DPP16 butterfly reductions within the 16-lane half-groups (pure VALU)
template<int CTRL>
__device__ inline float dpp_mov(float x) {
    int r = __builtin_amdgcn_update_dpp(0, __builtin_bit_cast(int, x),
                                        CTRL, 0xF, 0xF, true);
    return __builtin_bit_cast(float, r);
}
__device__ inline float rowmax16(float x) {
    x = fmaxf(x, dpp_mov<0xB1>(x));   // quad_perm [1,0,3,2]  (xor 1)
    x = fmaxf(x, dpp_mov<0x4E>(x));   // quad_perm [2,3,0,1]  (xor 2)
    x = fmaxf(x, dpp_mov<0x141>(x));  // row_half_mirror      (xor 4 group)
    x = fmaxf(x, dpp_mov<0x140>(x));  // row_mirror           (xor 8 group)
    return x;
}
__device__ inline float rowsum16(float x) {
    x += dpp_mov<0xB1>(x);
    x += dpp_mov<0x4E>(x);
    x += dpp_mov<0x141>(x);
    x += dpp_mov<0x140>(x);
    return x;
}

// ---------------------------------------------------------------- converts
__global__ __launch_bounds__(256)
void cvt_f32_to_bf16(const float* __restrict__ in, __bf16* __restrict__ out, size_t n) {
    size_t i = (size_t)blockIdx.x * blockDim.x + threadIdx.x;
    size_t stride = (size_t)gridDim.x * blockDim.x;
    for (; i < n; i += stride) out[i] = (__bf16)in[i];
}

// in[R][C] f32 -> out[C][R] bf16, 32x32 LDS tiles (both sides coalesced)
__global__ __launch_bounds__(256)
void cvt_transpose_bf16(const float* __restrict__ in, __bf16* __restrict__ out,
                        int R, int Ccols) {
    __shared__ __bf16 tile[32][33];
    const int bx = blockIdx.x * 32;           // col base
    const int by = blockIdx.y * 32;           // row base
    const int tx = threadIdx.x & 31;
    const int ty = threadIdx.x >> 5;          // 0..7
#pragma unroll
    for (int i = 0; i < 32; i += 8)
        tile[ty + i][tx] = (__bf16)in[(size_t)(by + ty + i) * Ccols + bx + tx];
    __syncthreads();
#pragma unroll
    for (int i = 0; i < 32; i += 8)
        out[(size_t)(bx + ty + i) * R + by + tx] = tile[tx][ty + i];
}

// ---------------------------------------------------------------- GEMM
// C[M,N] = A[M,K] @ BT[N,K]^T + bias[N]. 256 thr / 8 waves, 128x128 tile,
// K-step 32, double-buffered LDS fed by async global->LDS, 1 barrier/step.
// QKV=true: cols < 2C -> Cqk[M,2C] bf16 ; cols >= 2C (the V block) scattered
// into CvT[B,H,D,T] bf16 so attention can async-load V rows directly.
template<bool QKV>
__global__ __launch_bounds__(256)
void gemm_bf16_async(const __bf16* __restrict__ A, const __bf16* __restrict__ BT,
                     const float* __restrict__ bias,
                     float* __restrict__ Cf, __bf16* __restrict__ Cqk,
                     __bf16* __restrict__ CvT, int M, int N, int K) {
    __shared__ __bf16 sA[2][128 * 32];
    __shared__ __bf16 sB[2][128 * 32];
    const int t = threadIdx.x;
    const int wave = t >> 5, lane = t & 31;
    const int l16 = lane & 15, hi = lane >> 4;
    const int m0 = blockIdx.y * 128, n0 = blockIdx.x * 128;
    const int wm = (wave >> 1) * 32;
    const int wn = (wave & 1) * 64;

    // per-thread 32B slice of each 128x32 tile
    const int idx = t * 16, lrow = idx >> 5, lcol = idx & 31;
    const unsigned la[2] = { lds_off(&sA[0][lrow * 32 + lcol]),
                             lds_off(&sA[1][lrow * 32 + lcol]) };
    const unsigned lb[2] = { lds_off(&sB[0][lrow * 32 + lcol]),
                             lds_off(&sB[1][lrow * 32 + lcol]) };
    const __bf16* gA = A  + (size_t)(m0 + lrow) * K + lcol;
    const __bf16* gB = BT + (size_t)(n0 + lrow) * K + lcol;

    auto issue = [&](int kk, int buf) {
        async_cp16(la[buf],      gA + kk);
        async_cp16(la[buf] + 16, gA + kk + 8);
        async_cp16(lb[buf],      gB + kk);
        async_cp16(lb[buf] + 16, gB + kk + 8);
    };

    const v8f vzero = {0.f,0.f,0.f,0.f,0.f,0.f,0.f,0.f};
    v8f acc[2][4];
#pragma unroll
    for (int i = 0; i < 2; ++i)
#pragma unroll
        for (int j = 0; j < 4; ++j) acc[i][j] = vzero;

    const int steps = K / 32;
    issue(0, 0);
    for (int i = 0; i < steps; ++i) {
        const int buf = i & 1;
        wait_async0();           // own async writes to buf done
        __syncthreads();         // everyone's done (and reads of buf^1 done)
        if (i + 1 < steps) issue((i + 1) * 32, buf ^ 1);

        v16bf af[2], bfr[4];
#pragma unroll
        for (int mt = 0; mt < 2; ++mt)
            af[mt] = load_afrag(&sA[buf][(wm + 16 * mt + l16) * 32], hi);
#pragma unroll
        for (int nt = 0; nt < 4; ++nt)
            bfr[nt] = load_bfrag(&sB[buf][(wn + 16 * nt + l16) * 32], hi);
#pragma unroll
        for (int mt = 0; mt < 2; ++mt)
#pragma unroll
            for (int nt = 0; nt < 4; ++nt)
                acc[mt][nt] = wmma_bf16(af[mt], bfr[nt], acc[mt][nt]);
    }

    // epilogue: lane n = l16, reg r -> row r + 8*hi
#pragma unroll
    for (int mt = 0; mt < 2; ++mt)
#pragma unroll
        for (int nt = 0; nt < 4; ++nt) {
            const int col = n0 + wn + nt * 16 + l16;
            const float bv = bias[col];
#pragma unroll
            for (int r = 0; r < 8; ++r) {
                const int row = m0 + wm + mt * 16 + r + 8 * hi;
                const float v = acc[mt][nt][r] + bv;
                if (QKV) {
                    if (n0 < C2) {
                        Cqk[(size_t)row * C2 + col] = (__bf16)v;
                    } else {
                        const int hd = col - C2, hh = hd >> 6, dd = hd & 63;
                        const int bb = row >> 11, tt = row & (SEQ - 1);
                        CvT[((size_t)(bb * NHEAD + hh) * HDIM + dd) * SEQ + tt] = (__bf16)v;
                    }
                } else {
                    Cf[(size_t)row * N + col] = v;
                }
            }
        }
}

// ---------------------------------------------------------------- flash attention
// grid (T/64, H, B), block 128 (4 waves, 16 q-rows each). Key blocks of 32,
// double-buffered async K/V tiles, online softmax with DPP reductions.
__global__ __launch_bounds__(128)
void flash_attn_bf16(const __bf16* __restrict__ qk, const __bf16* __restrict__ vT,
                     __bf16* __restrict__ y) {
    __shared__ __bf16 sK[2][32 * 64];   // [key][d]
    __shared__ __bf16 sV[2][64 * 32];   // [d][key]
    __shared__ __bf16 sP[4 * 512];      // per-wave P staging

    const int tid = threadIdx.x;
    const int wave = tid >> 5, lane = tid & 31;
    const int l16 = lane & 15, hi = lane >> 4;
    const int b = blockIdx.z, h = blockIdx.y;
    const int qBlock = blockIdx.x * 64;
    const int qBase = qBlock + wave * 16;
    const size_t rowB = (size_t)b * SEQ;

    // per-thread async slices: K tile 32x64, V tile 64x32 (32B each)
    const int idx = tid * 16;
    const int krow = idx >> 6, dcol = idx & 63;
    const int dr = idx >> 5, kc = idx & 31;
    const __bf16* gK = qk + (rowB + krow) * C2 + CH + h * HDIM + dcol;
    const __bf16* gV = vT + ((size_t)(b * NHEAD + h) * HDIM + dr) * SEQ + kc;
    const unsigned lk[2] = { lds_off(&sK[0][krow * 64 + dcol]),
                             lds_off(&sK[1][krow * 64 + dcol]) };
    const unsigned lv[2] = { lds_off(&sV[0][dr * 32 + kc]),
                             lds_off(&sV[1][dr * 32 + kc]) };
    auto issue = [&](int kk, int buf) {
        async_cp16(lk[buf],      gK + (size_t)kk * C2);
        async_cp16(lk[buf] + 16, gK + (size_t)kk * C2 + 8);
        async_cp16(lv[buf],      gV + kk);
        async_cp16(lv[buf] + 16, gV + kk + 8);
    };

    // Q fragments (16 x 64), two K=32 chunks
    v16bf aq[2];
#pragma unroll
    for (int c = 0; c < 2; ++c)
        aq[c] = load_afrag(qk + (rowB + qBase + l16) * C2 + h * HDIM + c * 32, hi);

    const v8f vzero = {0.f,0.f,0.f,0.f,0.f,0.f,0.f,0.f};
    v8f o[4] = {vzero, vzero, vzero, vzero};
    float m[8], l[8];
#pragma unroll
    for (int r = 0; r < 8; ++r) { m[r] = -1e30f; l[r] = 0.f; }

    const int nkb = qBlock / 32 + 2;     // key blocks covering keys <= qBlock+63
    issue(0, 0);
    for (int kb = 0; kb < nkb; ++kb) {
        const int kk = kb * 32;
        const int buf = kb & 1;
        wait_async0();
        __syncthreads();
        if (kb + 1 < nkb) issue((kb + 1) * 32, buf ^ 1);

        if (kk <= qBase + 15) {          // wave-uniform causal skip
            v8f s[2] = {vzero, vzero};
#pragma unroll
            for (int nt = 0; nt < 2; ++nt)
#pragma unroll
                for (int c = 0; c < 2; ++c) {
                    v16bf bk = load_bfrag(&sK[buf][(nt * 16 + l16) * 64 + c * 32], hi);
                    s[nt] = wmma_bf16(aq[c], bk, s[nt]);
                }

            float p0[8], p1[8];
#pragma unroll
            for (int r = 0; r < 8; ++r) {
                const int q = qBase + r + 8 * hi;
                float v0 = s[0][r] * 0.125f;
                float v1 = s[1][r] * 0.125f;
                if (kk + l16 > q)      v0 = -1e30f;
                if (kk + 16 + l16 > q) v1 = -1e30f;
                const float mx = rowmax16(fmaxf(v0, v1));
                const float mn = fmaxf(m[r], mx);
                const float a  = __expf(m[r] - mn);
                p0[r] = __expf(v0 - mn);
                p1[r] = __expf(v1 - mn);
                l[r] = l[r] * a + rowsum16(p0[r] + p1[r]);
                m[r] = mn;
#pragma unroll
                for (int dt = 0; dt < 4; ++dt) o[dt][r] *= a;
            }

            // restage P (C layout -> A layout) through per-wave LDS
            __bf16* pw = sP + wave * 512;
#pragma unroll
            for (int r = 0; r < 8; ++r) {
                pw[(r + 8 * hi) * 32 + l16]      = (__bf16)p0[r];
                pw[(r + 8 * hi) * 32 + 16 + l16] = (__bf16)p1[r];
            }
            asm volatile("s_wait_dscnt 0" ::: "memory");   // wave-local LDS drain

            v16bf pa = load_afrag(pw + l16 * 32, hi);
#pragma unroll
            for (int dt = 0; dt < 4; ++dt) {
                v16bf bv = load_bfrag(&sV[buf][(dt * 16 + l16) * 32], hi);
                o[dt] = wmma_bf16(pa, bv, o[dt]);
            }
        }
    }

    // normalize and store y[b, q, h*64 + d] bf16
#pragma unroll
    for (int r = 0; r < 8; ++r) {
        const float inv = 1.0f / l[r];
        const int q = qBase + r + 8 * hi;
#pragma unroll
        for (int dt = 0; dt < 4; ++dt)
            y[(rowB + q) * CH + h * HDIM + dt * 16 + l16] = (__bf16)(o[dt][r] * inv);
    }
}

// ---------------------------------------------------------------- launch
extern "C" void kernel_launch(void* const* d_in, const int* in_sizes, int n_in,
                              void* d_out, int out_size, void* d_ws, size_t ws_size,
                              hipStream_t stream) {
    const float* x      = (const float*)d_in[0];
    const float* W_attn = (const float*)d_in[1];
    const float* b_attn = (const float*)d_in[2];
    const float* W_proj = (const float*)d_in[3];
    const float* b_proj = (const float*)d_in[4];
    float* out = (float*)d_out;

    const size_t M = (size_t)BATCH * SEQ;   // 8192
    size_t off = 0;
    auto carve = [&](size_t bytes) {
        void* p = (char*)d_ws + off;
        off += (bytes + 255) & ~(size_t)255;
        return p;
    };
    __bf16* xb   = (__bf16*)carve(M * CH * 2);            // x bf16
    __bf16* WaT  = (__bf16*)carve((size_t)C3 * CH * 2);   // W_attn^T [3C][C]
    __bf16* WpT  = (__bf16*)carve((size_t)CH * CH * 2);   // W_proj^T [C][C]
    __bf16* qkb  = (__bf16*)carve(M * C2 * 2);            // Q|K [M][2C]
    __bf16* vTb  = (__bf16*)carve(M * CH * 2);            // V^T [B][H][D][T]
    __bf16* yb   = (__bf16*)carve(M * CH * 2);            // attention out bf16

    cvt_f32_to_bf16<<<4096, 256, 0, stream>>>(x, xb, M * CH);
    cvt_transpose_bf16<<<dim3(C3 / 32, CH / 32), 256, 0, stream>>>(W_attn, WaT, CH, C3);
    cvt_transpose_bf16<<<dim3(CH / 32, CH / 32), 256, 0, stream>>>(W_proj, WpT, CH, CH);

    // QKV projection: [8192,1024] @ [1024,3072] + b_attn -> qkb / vTb
    gemm_bf16_async<true><<<dim3(C3 / 128, M / 128), 256, 0, stream>>>(
        xb, WaT, b_attn, nullptr, qkb, vTb, (int)M, C3, CH);

    // causal flash attention
    flash_attn_bf16<<<dim3(SEQ / 64, NHEAD, BATCH), 128, 0, stream>>>(qkb, vTb, yb);

    // output projection: [8192,1024] @ [1024,1024] + b_proj -> f32 out
    gemm_bf16_async<false><<<dim3(CH / 128, M / 128), 256, 0, stream>>>(
        yb, WpT, b_proj, out, nullptr, nullptr, (int)M, CH, CH);
}
